// AutoformerLayer_44006234915544
// MI455X (gfx1250) — matrix operand order repaired and checked by reading, not compiled, not run on hardware
//
#include <hip/hip_runtime.h>
#include <hip/hip_bf16.h>
#include <math.h>

// ---------------------------------------------------------------------------
// Autoformer layer for MI455X (gfx1250): all GEMMs via v_wmma_f32_16x16x32_bf16
// (fp32 accumulate). Compute-bound (~210 GFLOP of GEMM); weights pre-converted
// to bf16 transposed [N][K] so WMMA B-fragments are contiguous 16B loads.
// L2 (192 MB) holds all weight matrices + activations between kernels.
// ---------------------------------------------------------------------------

typedef __bf16 bf16_t;
typedef __attribute__((ext_vector_type(16))) __bf16 v16bf;
typedef __attribute__((ext_vector_type(8)))  __bf16 v8bf;
typedef __attribute__((ext_vector_type(8)))  float  v8f;

#define DIMD   512
#define DFF    2048
#define HEADS  8
#define HD     64
#define BATCH  8
#define SEQ    4096
#define TOK    (BATCH*SEQ)

// ---------------- WMMA fragment loads (bf16, 16x16x32) ----------------------
// A (16x32, row-major [M,K]): lanes 0-15 row M=lane, K={0..7,16..23};
//                             lanes 16-31 row M=lane-16, K={8..15,24..31}.
__device__ __forceinline__ v16bf load_a_frag(const bf16_t* A, int lda, int m0,
                                             int k0, int lane) {
  int row = m0 + (lane & 15);
  int kb  = k0 + ((lane >> 4) << 3);
  const bf16_t* p = A + (size_t)row * lda + kb;
  v8bf lo = *reinterpret_cast<const v8bf*>(p);
  v8bf hi = *reinterpret_cast<const v8bf*>(p + 16);
  v16bf r;
#pragma unroll
  for (int i = 0; i < 8; ++i) { r[i] = lo[i]; r[i + 8] = hi[i]; }
  return r;
}

// B (32x16) from Bt stored [N,K] row-major: lane holds column N=lane&15,
// lanes 0-15 K={0..7,16..23}, lanes 16-31 K={8..15,24..31}.
__device__ __forceinline__ v16bf load_b_frag(const bf16_t* Bt, int ldb, int n0,
                                             int k0, int lane) {
  int colN = n0 + (lane & 15);
  int kb   = k0 + ((lane >> 4) << 3);
  const bf16_t* p = Bt + (size_t)colN * ldb + kb;
  v8bf lo = *reinterpret_cast<const v8bf*>(p);
  v8bf hi = *reinterpret_cast<const v8bf*>(p + 16);
  v16bf r;
#pragma unroll
  for (int i = 0; i < 8; ++i) { r[i] = lo[i]; r[i + 8] = hi[i]; }
  return r;
}

__device__ __forceinline__ v8f wmma_bf16(v16bf a, v16bf b, v8f c) {
  return __builtin_amdgcn_wmma_f32_16x16x32_bf16(false, a, false, b,
                                                 (short)0, c, false, false);
}

// ------------------------------- epilogues ----------------------------------
__device__ __forceinline__ float elu_p1(float v) {
  return v > 0.f ? v + 1.f : __expf(v);
}

struct EpiQKV {  // fused QKV: bias + elu+1 (q,k), scatter to layouts
  const float *bq, *bk, *bv;
  __hip_bfloat16 *Qb;  // [b,h,n,64]
  __hip_bfloat16 *Kt;  // [b,h,64,n]
  __hip_bfloat16 *Vt;  // [b,h,64,n]
  __device__ void operator()(int, int row, int col, float v) const {
    int b = row >> 12, t = row & 4095;
    if (col < DIMD) {
      int h = col >> 6, d = col & 63;
      float q = elu_p1(v + bq[col]);
      Qb[((size_t)(b * HEADS + h) * SEQ + t) * HD + d] = __float2bfloat16(q);
    } else if (col < 2 * DIMD) {
      int c = col - DIMD, h = c >> 6, d = c & 63;
      float k = elu_p1(v + bk[c]);
      Kt[((size_t)(b * HEADS + h) * HD + d) * SEQ + t] = __float2bfloat16(k);
    } else {
      int c = col - 2 * DIMD, h = c >> 6, f = c & 63;
      Vt[((size_t)(b * HEADS + h) * HD + f) * SEQ + t] =
          __float2bfloat16(v + bv[c]);
    }
  }
};

struct EpiOutPre {  // attention out (pre-Wo) -> token-major [tok,512]
  __hip_bfloat16* Ob;
  __device__ void operator()(int bz, int row, int col, float v) const {
    int b = bz >> 3, h = bz & 7;
    Ob[((size_t)(b * SEQ + row)) * DIMD + h * HD + col] = __float2bfloat16(v);
  }
};

struct EpiBiasAddF32 {  // S[row,col] += acc + bias  (residual accumulate)
  const float* bias;
  float* S;
  __device__ void operator()(int, int row, int col, float v) const {
    S[(size_t)row * DIMD + col] += v + bias[col];
  }
};

struct EpiBiasGelu {  // exact-erf GELU -> bf16 hidden
  const float* bias;
  __hip_bfloat16* H;
  __device__ void operator()(int, int row, int col, float v) const {
    v += bias[col];
    float g = 0.5f * v * (1.f + erff(v * 0.70710678118654752f));
    H[(size_t)row * DFF + col] = __float2bfloat16(g);
  }
};

// --------------------------- generic bf16 GEMM ------------------------------
// C[M,N] = A[M,K] * B[K,N]; A row-major bf16, Bt = B^T stored [N,K] bf16.
// Block: WGM*WGN waves (32 lanes), wave computes WMT x WNT 16x16 tiles.
template <class Epi, int WMT, int WNT, int WGM, int WGN>
__global__ __launch_bounds__(32 * WGM * WGN) void gemm_bf16_k(
    const bf16_t* __restrict__ A, int lda, long long strideAz,
    const bf16_t* __restrict__ Bt, int ldb, long long strideBz, int K,
    Epi epi) {
  constexpr int BM = 16 * WMT * WGM;
  constexpr int BN = 16 * WNT * WGN;
  const int wave = threadIdx.x >> 5, lane = threadIdx.x & 31;
  const int wm = wave % WGM, wn = wave / WGM;
  const int bz = blockIdx.z;
  A  += (long long)bz * strideAz;
  Bt += (long long)bz * strideBz;
  const int m_base = blockIdx.y * BM + wm * 16 * WMT;
  const int n_base = blockIdx.x * BN + wn * 16 * WNT;

  v8f acc[WMT][WNT];
#pragma unroll
  for (int mi = 0; mi < WMT; ++mi)
#pragma unroll
    for (int ni = 0; ni < WNT; ++ni)
#pragma unroll
      for (int e = 0; e < 8; ++e) acc[mi][ni][e] = 0.f;

  for (int k0 = 0; k0 < K; k0 += 32) {
    v16bf afr[WMT], bfr[WNT];
#pragma unroll
    for (int mi = 0; mi < WMT; ++mi)
      afr[mi] = load_a_frag(A, lda, m_base + 16 * mi, k0, lane);
#pragma unroll
    for (int ni = 0; ni < WNT; ++ni)
      bfr[ni] = load_b_frag(Bt, ldb, n_base + 16 * ni, k0, lane);
    // hint next A slab into cache (global_prefetch_b8)
    __builtin_prefetch(A + (size_t)(m_base + (lane & 15)) * lda + k0 + 32, 0, 1);
#pragma unroll
    for (int mi = 0; mi < WMT; ++mi)
#pragma unroll
      for (int ni = 0; ni < WNT; ++ni)
        acc[mi][ni] = wmma_bf16(afr[mi], bfr[ni], acc[mi][ni]);
  }

#pragma unroll
  for (int mi = 0; mi < WMT; ++mi)
#pragma unroll
    for (int ni = 0; ni < WNT; ++ni)
#pragma unroll
      for (int r = 0; r < 8; ++r) {
        int row = m_base + 16 * mi + r + ((lane >> 4) << 3);
        int col = n_base + 16 * ni + (lane & 15);
        epi(bz, row, col, acc[mi][ni][r]);
      }
}

// ------------- kv state GEMM: per (b,h)  kv^T[f,d] = (K^T V)^T --------------
// A = Kt[bh] (64 x 4096), Bt = Vt[bh] (64 x 4096). 8 waves split K=4096,
// LDS ds_add_f32 cross-wave reduction, write kv transposed (bf16).
__global__ __launch_bounds__(256) void gemm_kv_k(const bf16_t* __restrict__ Kt,
                                                 const bf16_t* __restrict__ Vt,
                                                 __hip_bfloat16* __restrict__ kvT) {
  __shared__ float lds[HD * HD];
  const int bh = blockIdx.x;
  const bf16_t* Ab = Kt + (size_t)bh * HD * SEQ;
  const bf16_t* Bb = Vt + (size_t)bh * HD * SEQ;
  const int wave = threadIdx.x >> 5, lane = threadIdx.x & 31;
  for (int i = threadIdx.x; i < HD * HD; i += 256) lds[i] = 0.f;
  __syncthreads();

  const int mhalf = (wave & 1) * 32;       // two M halves
  const int kbase = (wave >> 1) * 1024;    // four K chunks
  v8f acc[2][4];
#pragma unroll
  for (int mi = 0; mi < 2; ++mi)
#pragma unroll
    for (int ni = 0; ni < 4; ++ni)
#pragma unroll
      for (int e = 0; e < 8; ++e) acc[mi][ni][e] = 0.f;

  for (int k0 = kbase; k0 < kbase + 1024; k0 += 32) {
    v16bf a0 = load_a_frag(Ab, SEQ, mhalf, k0, lane);
    v16bf a1 = load_a_frag(Ab, SEQ, mhalf + 16, k0, lane);
    v16bf b[4];
#pragma unroll
    for (int ni = 0; ni < 4; ++ni) b[ni] = load_b_frag(Bb, SEQ, ni * 16, k0, lane);
#pragma unroll
    for (int ni = 0; ni < 4; ++ni) {
      acc[0][ni] = wmma_bf16(a0, b[ni], acc[0][ni]);
      acc[1][ni] = wmma_bf16(a1, b[ni], acc[1][ni]);
    }
  }
#pragma unroll
  for (int mi = 0; mi < 2; ++mi)
#pragma unroll
    for (int ni = 0; ni < 4; ++ni)
#pragma unroll
      for (int r = 0; r < 8; ++r) {
        int row = mhalf + 16 * mi + r + ((lane >> 4) << 3);
        int col = 16 * ni + (lane & 15);
        atomicAdd(&lds[row * HD + col], acc[mi][ni][r]);  // ds_add_f32
      }
  __syncthreads();
  for (int j = threadIdx.x; j < HD * HD; j += 256) {
    int f = j >> 6, d = j & 63;  // kvT[f,d] = C[d,f]
    kvT[(size_t)bh * HD * HD + j] = __float2bfloat16(lds[d * HD + f]);
  }
}

// -------------------- decomposition + LayerNorm kernels ---------------------
__global__ __launch_bounds__(256) void decomp_ln1_k(
    const float* __restrict__ x, const float* __restrict__ g,
    const float* __restrict__ bb, float* __restrict__ S,
    __hip_bfloat16* __restrict__ Xb) {
  const int wave = threadIdx.x >> 5, lane = threadIdx.x & 31;
  const int t = blockIdx.x * 8 + wave;           // global token
  const int n = t & (SEQ - 1);
  const float* xc = x + (size_t)t * DIMD;
  float s[16];
  float sum = 0.f, sq = 0.f;
#pragma unroll
  for (int j = 0; j < 4; ++j) {
    int col = lane * 4 + j * 128;
    float4 c = *(const float4*)(xc + col);
    float4 p = make_float4(0.f, 0.f, 0.f, 0.f);
    float4 q = make_float4(0.f, 0.f, 0.f, 0.f);
    if (n > 0)       p = *(const float4*)(xc - DIMD + col);
    if (n < SEQ - 1) q = *(const float4*)(xc + DIMD + col);
    float sv[4];
    sv[0] = c.x - (p.x + c.x + q.x) * (1.f / 3.f);
    sv[1] = c.y - (p.y + c.y + q.y) * (1.f / 3.f);
    sv[2] = c.z - (p.z + c.z + q.z) * (1.f / 3.f);
    sv[3] = c.w - (p.w + c.w + q.w) * (1.f / 3.f);
    *(float4*)(S + (size_t)t * DIMD + col) =
        make_float4(sv[0], sv[1], sv[2], sv[3]);
#pragma unroll
    for (int e = 0; e < 4; ++e) {
      s[j * 4 + e] = sv[e];
      sum += sv[e];
      sq += sv[e] * sv[e];
    }
  }
#pragma unroll
  for (int off = 16; off; off >>= 1) {
    sum += __shfl_xor(sum, off);
    sq  += __shfl_xor(sq, off);
  }
  float mean = sum * (1.f / DIMD);
  float var  = sq * (1.f / DIMD) - mean * mean;
  float rstd = rsqrtf(var + 1e-5f);
#pragma unroll
  for (int j = 0; j < 4; ++j) {
    int col = lane * 4 + j * 128;
    float4 gv = *(const float4*)(g + col);
    float4 bv = *(const float4*)(bb + col);
    Xb[(size_t)t * DIMD + col + 0] = __float2bfloat16((s[j*4+0]-mean)*rstd*gv.x + bv.x);
    Xb[(size_t)t * DIMD + col + 1] = __float2bfloat16((s[j*4+1]-mean)*rstd*gv.y + bv.y);
    Xb[(size_t)t * DIMD + col + 2] = __float2bfloat16((s[j*4+2]-mean)*rstd*gv.z + bv.z);
    Xb[(size_t)t * DIMD + col + 3] = __float2bfloat16((s[j*4+3]-mean)*rstd*gv.w + bv.w);
  }
}

__global__ __launch_bounds__(256) void ln2_k(const float* __restrict__ S,
                                             const float* __restrict__ g,
                                             const float* __restrict__ bb,
                                             __hip_bfloat16* __restrict__ Xb) {
  const int wave = threadIdx.x >> 5, lane = threadIdx.x & 31;
  const int t = blockIdx.x * 8 + wave;
  const float* sr = S + (size_t)t * DIMD;
  float s[16];
  float sum = 0.f, sq = 0.f;
#pragma unroll
  for (int j = 0; j < 4; ++j) {
    int col = lane * 4 + j * 128;
    float4 v = *(const float4*)(sr + col);
    s[j*4+0] = v.x; s[j*4+1] = v.y; s[j*4+2] = v.z; s[j*4+3] = v.w;
    sum += v.x + v.y + v.z + v.w;
    sq  += v.x*v.x + v.y*v.y + v.z*v.z + v.w*v.w;
  }
#pragma unroll
  for (int off = 16; off; off >>= 1) {
    sum += __shfl_xor(sum, off);
    sq  += __shfl_xor(sq, off);
  }
  float mean = sum * (1.f / DIMD);
  float rstd = rsqrtf(sq * (1.f / DIMD) - mean * mean + 1e-5f);
#pragma unroll
  for (int j = 0; j < 4; ++j) {
    int col = lane * 4 + j * 128;
    float4 gv = *(const float4*)(g + col);
    float4 bv = *(const float4*)(bb + col);
    Xb[(size_t)t * DIMD + col + 0] = __float2bfloat16((s[j*4+0]-mean)*rstd*gv.x + bv.x);
    Xb[(size_t)t * DIMD + col + 1] = __float2bfloat16((s[j*4+1]-mean)*rstd*gv.y + bv.y);
    Xb[(size_t)t * DIMD + col + 2] = __float2bfloat16((s[j*4+2]-mean)*rstd*gv.z + bv.z);
    Xb[(size_t)t * DIMD + col + 3] = __float2bfloat16((s[j*4+3]-mean)*rstd*gv.w + bv.w);
  }
}

// out = S + trend(x)  (trend recomputed: cheap, avoids 128 MB of traffic)
__global__ __launch_bounds__(256) void final_k(const float* __restrict__ x,
                                               const float* __restrict__ S,
                                               float* __restrict__ out) {
  int i = blockIdx.x * 256 + threadIdx.x;  // one float4 per thread
  int t = i >> 7, c4 = i & 127;
  int n = t & (SEQ - 1);
  const float* xc = x + (size_t)t * DIMD + c4 * 4;
  float4 c = *(const float4*)xc;
  float4 p = make_float4(0.f, 0.f, 0.f, 0.f);
  float4 q = make_float4(0.f, 0.f, 0.f, 0.f);
  if (n > 0)       p = *(const float4*)(xc - DIMD);
  if (n < SEQ - 1) q = *(const float4*)(xc + DIMD);
  float4 sv = *(const float4*)(S + (size_t)t * DIMD + c4 * 4);
  float4 o;
  o.x = sv.x + (p.x + c.x + q.x) * (1.f / 3.f);
  o.y = sv.y + (p.y + c.y + q.y) * (1.f / 3.f);
  o.z = sv.z + (p.z + c.z + q.z) * (1.f / 3.f);
  o.w = sv.w + (p.w + c.w + q.w) * (1.f / 3.f);
  *(float4*)(out + (size_t)t * DIMD + c4 * 4) = o;
}

// weight transpose + f32->bf16:  dst[n*K+k] = src[k*N+n]
__global__ __launch_bounds__(256) void wconv_k(const float* __restrict__ src,
                                               __hip_bfloat16* __restrict__ dst,
                                               int K, int N) {
  long long idx = (long long)blockIdx.x * 256 + threadIdx.x;
  if (idx >= (long long)K * N) return;
  int n = (int)(idx / K), k = (int)(idx % K);
  dst[idx] = __float2bfloat16(src[(size_t)k * N + n]);
}

// ------------------------------- launcher -----------------------------------
extern "C" void kernel_launch(void* const* d_in, const int* in_sizes, int n_in,
                              void* d_out, int out_size, void* d_ws,
                              size_t ws_size, hipStream_t stream) {
  const float* x   = (const float*)d_in[0];
  const float* Wq  = (const float*)d_in[1];
  const float* bq  = (const float*)d_in[2];
  const float* Wk  = (const float*)d_in[3];
  const float* bk  = (const float*)d_in[4];
  const float* Wv  = (const float*)d_in[5];
  const float* bv  = (const float*)d_in[6];
  const float* Wo  = (const float*)d_in[7];
  const float* bo  = (const float*)d_in[8];
  const float* g1  = (const float*)d_in[9];
  const float* b1  = (const float*)d_in[10];
  const float* g2  = (const float*)d_in[11];
  const float* b2  = (const float*)d_in[12];
  const float* Wf1 = (const float*)d_in[13];
  const float* bf1 = (const float*)d_in[14];
  const float* Wf2 = (const float*)d_in[15];
  const float* bf2 = (const float*)d_in[16];
  float* out = (float*)d_out;

  char* ws = (char*)d_ws;
  const size_t OFF_S    = 0;                                   // f32 [TOK,512]
  const size_t OFF_XB   = OFF_S    + (size_t)TOK * DIMD * 4;   // bf16 LN out
  const size_t OFF_QB   = OFF_XB   + (size_t)TOK * DIMD * 2;   // bf16 [b,h,n,64]
  const size_t OFF_KT   = OFF_QB   + (size_t)TOK * DIMD * 2;   // bf16 [b,h,64,n]
  const size_t OFF_VT   = OFF_KT   + (size_t)TOK * DIMD * 2;   // bf16 [b,h,64,n]
  const size_t OFF_OB   = OFF_VT   + (size_t)TOK * DIMD * 2;   // bf16 [tok,512]
  const size_t OFF_H    = OFF_OB   + (size_t)TOK * DIMD * 2;   // bf16 [tok,2048]
  const size_t OFF_KV   = OFF_H    + (size_t)TOK * DFF  * 2;   // bf16 kv^T
  const size_t OFF_WQKV = OFF_KV   + (size_t)BATCH*HEADS*HD*HD*2;
  const size_t OFF_WO   = OFF_WQKV + (size_t)3 * DIMD * DIMD * 2;
  const size_t OFF_WF1  = OFF_WO   + (size_t)DIMD * DIMD * 2;
  const size_t OFF_WF2  = OFF_WF1  + (size_t)DIMD * DFF * 2;

  float*          S    = (float*)(ws + OFF_S);
  __hip_bfloat16* Xb   = (__hip_bfloat16*)(ws + OFF_XB);
  __hip_bfloat16* Qb   = (__hip_bfloat16*)(ws + OFF_QB);
  __hip_bfloat16* Ktb  = (__hip_bfloat16*)(ws + OFF_KT);
  __hip_bfloat16* Vtb  = (__hip_bfloat16*)(ws + OFF_VT);
  __hip_bfloat16* Ob   = (__hip_bfloat16*)(ws + OFF_OB);
  __hip_bfloat16* H    = (__hip_bfloat16*)(ws + OFF_H);
  __hip_bfloat16* kvT  = (__hip_bfloat16*)(ws + OFF_KV);
  __hip_bfloat16* Wqkvt= (__hip_bfloat16*)(ws + OFF_WQKV);
  __hip_bfloat16* Wot  = (__hip_bfloat16*)(ws + OFF_WO);
  __hip_bfloat16* Wf1t = (__hip_bfloat16*)(ws + OFF_WF1);
  __hip_bfloat16* Wf2t = (__hip_bfloat16*)(ws + OFF_WF2);

  // 1) weights -> bf16 transposed [N,K]
  wconv_k<<<(DIMD*DIMD + 255) / 256, 256, 0, stream>>>(Wq, Wqkvt, DIMD, DIMD);
  wconv_k<<<(DIMD*DIMD + 255) / 256, 256, 0, stream>>>(Wk, Wqkvt + DIMD*DIMD, DIMD, DIMD);
  wconv_k<<<(DIMD*DIMD + 255) / 256, 256, 0, stream>>>(Wv, Wqkvt + 2*DIMD*DIMD, DIMD, DIMD);
  wconv_k<<<(DIMD*DIMD + 255) / 256, 256, 0, stream>>>(Wo, Wot, DIMD, DIMD);
  wconv_k<<<(DIMD*DFF  + 255) / 256, 256, 0, stream>>>(Wf1, Wf1t, DIMD, DFF);
  wconv_k<<<(DFF*DIMD  + 255) / 256, 256, 0, stream>>>(Wf2, Wf2t, DFF, DIMD);

  // 2) trend/seasonal + LN1 -> Xb (bf16), S = seasonal (f32 residual acc)
  decomp_ln1_k<<<TOK / 8, 256, 0, stream>>>(x, g1, b1, S, Xb);

  // 3) fused QKV projection: [32768,512] x [512,1536]
  gemm_bf16_k<EpiQKV, 2, 4, 4, 2><<<dim3(12, 256, 1), 256, 0, stream>>>(
      (const bf16_t*)Xb, DIMD, 0, (const bf16_t*)Wqkvt, DIMD, 0, DIMD,
      EpiQKV{bq, bk, bv, Qb, Ktb, Vtb});

  // 4) kv state per (b,h): 64x64, K=4096 reduction
  gemm_kv_k<<<BATCH * HEADS, 256, 0, stream>>>((const bf16_t*)Ktb,
                                               (const bf16_t*)Vtb, kvT);

  // 5) out_pre = Q @ kv per (b,h): M=4096, N=64, K=64
  gemm_bf16_k<EpiOutPre, 1, 4, 8, 1><<<dim3(1, SEQ / 128, BATCH * HEADS), 256, 0,
                                       stream>>>(
      (const bf16_t*)Qb, HD, (long long)SEQ * HD, (const bf16_t*)kvT, HD,
      (long long)HD * HD, HD, EpiOutPre{Ob});

  // 6) O projection, accumulate into seasonal: S += Ob@Wo + bo
  gemm_bf16_k<EpiBiasAddF32, 2, 4, 4, 2><<<dim3(4, 256, 1), 256, 0, stream>>>(
      (const bf16_t*)Ob, DIMD, 0, (const bf16_t*)Wot, DIMD, 0, DIMD,
      EpiBiasAddF32{bo, S});

  // 7) LN2: S -> Xb
  ln2_k<<<TOK / 8, 256, 0, stream>>>(S, g2, b2, Xb);

  // 8) FFN1 + GELU: [32768,512] x [512,2048]
  gemm_bf16_k<EpiBiasGelu, 2, 4, 4, 2><<<dim3(16, 256, 1), 256, 0, stream>>>(
      (const bf16_t*)Xb, DIMD, 0, (const bf16_t*)Wf1t, DIMD, 0, DIMD,
      EpiBiasGelu{bf1, H});

  // 9) FFN2, accumulate: S += H@Wf2 + bf2
  gemm_bf16_k<EpiBiasAddF32, 2, 4, 4, 2><<<dim3(4, 256, 1), 256, 0, stream>>>(
      (const bf16_t*)H, DFF, 0, (const bf16_t*)Wf2t, DFF, 0, DFF,
      EpiBiasAddF32{bf2, S});

  // 10) out = S + trend(x)
  final_k<<<(TOK * DIMD / 4) / 256, 256, 0, stream>>>(x, S, out);
}
